// BiLSTM__13683765805742
// MI455X (gfx1250) — matrix operand order repaired and checked by reading, not compiled, not run on hardware
//
#include <hip/hip_runtime.h>

// ---------- types ----------
typedef __attribute__((ext_vector_type(16))) __bf16 v16bf;
typedef __attribute__((ext_vector_type(8)))  float  v8f;
typedef __attribute__((ext_vector_type(4)))  unsigned int u32x4;
typedef __attribute__((ext_vector_type(4)))  float  f32x4;

union Frag {
  v16bf bf;
  u32x4 u4[2];
  unsigned int u[8];
  unsigned short s[16];
};

// ---------- helpers ----------
__device__ __forceinline__ unsigned short f2bf(float f) {
  unsigned int u = __float_as_uint(f);
  u += 0x7FFFu + ((u >> 16) & 1u);          // round-to-nearest-even
  return (unsigned short)(u >> 16);
}
__device__ __forceinline__ unsigned int pack2bf(float lo, float hi) {
  return (unsigned int)f2bf(lo) | ((unsigned int)f2bf(hi) << 16);
}
__device__ __forceinline__ float sigm(float x) {
  return 1.0f / (1.0f + __expf(-x));
}
__device__ __forceinline__ float tanh_f(float x) {
  float e = __expf(2.0f * x);
  return 1.0f - 2.0f / (e + 1.0f);
}
__device__ __forceinline__ float swz8(float x) {   // lane xor-8 exchange
  return __int_as_float(__builtin_amdgcn_ds_swizzle(__float_as_int(x), 0x201F));
}
__device__ __forceinline__ v8f vzero8() {
  v8f z = {0.f,0.f,0.f,0.f,0.f,0.f,0.f,0.f};
  return z;
}
__device__ __forceinline__ v8f wmma_bf16(v16bf a, v16bf b, v8f c) {
  return __builtin_amdgcn_wmma_f32_16x16x32_bf16(false, a, false, b, (short)0, c, false, false);
}
// gate-column permutation: p -> original column.
// p bits: [9:5]=wave w, [4:3]=gate (0:i 1:g 2:f 3:o), [2:0]=unit-in-wave
__device__ __forceinline__ int permcol(int p) {
  return ((p >> 3) & 3) * 256 + (p >> 5) * 8 + (p & 7);
}

// ---------- workspace layout (bytes) ----------
#define WXDW 131072u  // dwords per direction for one 256x1024 bf16 frag matrix
static constexpr size_t OFF_WXF  = 0;                         // 2 * 512 KB
static constexpr size_t OFF_WHF  = 2ull * WXDW * 4;           // 2 * 512 KB
static constexpr size_t OFF_WFC  = OFF_WHF + 2ull * WXDW * 4; // 256 KB
static constexpr size_t OFF_BIAS = OFF_WFC + 65536ull * 4;    // 8 KB
static constexpr size_t OFF_HFIN = OFF_BIAS + 2048ull * 4;    // 64 KB
static constexpr size_t OFF_GATE = OFF_HFIN + 2ull * 64 * 256 * 2; // 128 MB bf16
// total ~130.3 MB
// gates fragment layout: [dir][t][m 4][nt 64][lane 32][4 dwords (8 bf16)]
// dword offset of frag(dir,t,m,nt) = (((dir*512+t)*4+m)*64+nt)*128

// =====================================================================
// K0: convert fp32 weights -> bf16 WMMA B-fragment buffers (+ permuted bias)
// Fragment layout: [frag][lane 32][8 dwords]; frag = nt*NK + kt
// B frag element 2v(+1) of lane l = W[K = kt*32 + ((l&16)?8:0) + 2v + ((v&4)?8:0) (+1)][col]
// =====================================================================
__global__ void prep_kernel(const float* __restrict__ Wf, const float* __restrict__ bF,
                            const float* __restrict__ Wb, const float* __restrict__ bB,
                            const float* __restrict__ Wfc,
                            unsigned int* __restrict__ wxf, unsigned int* __restrict__ whf,
                            unsigned int* __restrict__ wfcf, float* __restrict__ biasp) {
  int id = blockIdx.x * 256 + threadIdx.x;
  if (id < (int)(2 * WXDW)) {                       // W_x fragments (rows 0..255)
    int dir = id / WXDW, r = id % WXDW;
    int f = r >> 8, lane = (r >> 3) & 31, v = r & 7;
    int nt = f >> 3, kt = f & 7;
    int col = permcol(nt * 16 + (lane & 15));
    int k = kt * 32 + ((lane & 16) ? 8 : 0) + 2 * v + ((v & 4) ? 8 : 0);
    const float* W = dir ? Wb : Wf;
    wxf[id] = pack2bf(W[(size_t)k * 1024 + col], W[(size_t)(k + 1) * 1024 + col]);
  } else if (id < (int)(4 * WXDW)) {                // W_h fragments (rows 256..511)
    int id2 = id - 2 * WXDW;
    int dir = id2 / WXDW, r = id2 % WXDW;
    int f = r >> 8, lane = (r >> 3) & 31, v = r & 7;
    int nt = f >> 3, kt = f & 7;
    int col = permcol(nt * 16 + (lane & 15));
    int k = 256 + kt * 32 + ((lane & 16) ? 8 : 0) + 2 * v + ((v & 4) ? 8 : 0);
    const float* W = dir ? Wb : Wf;
    whf[id2] = pack2bf(W[(size_t)k * 1024 + col], W[(size_t)(k + 1) * 1024 + col]);
  } else if (id < (int)(4 * WXDW + 65536)) {        // W_fc fragments (no permutation)
    int r = id - 4 * WXDW;
    int f = r >> 8, lane = (r >> 3) & 31, v = r & 7;
    int nt = f >> 4, kt = f & 15;
    int col = nt * 16 + (lane & 15);
    int k = kt * 32 + ((lane & 16) ? 8 : 0) + 2 * v + ((v & 4) ? 8 : 0);
    wfcf[r] = pack2bf(Wfc[(size_t)k * 256 + col], Wfc[(size_t)(k + 1) * 256 + col]);
  } else if (id < (int)(4 * WXDW + 65536 + 2048)) { // permuted bias (fp32)
    int r = id - (4 * WXDW + 65536);
    int dir = r >> 10, p = r & 1023;
    biasp[r] = (dir ? bB : bF)[permcol(p)];
  }
}

// =====================================================================
// K1: gates_x = bf16( x_src @ W_x + b ), written in C-fragment layout so the
// recurrent kernel reads it with 8 coalesced b128 loads per wave per step.
// dir==1 uses tf.reverse_sequence indexing per batch row.
// Grid: (512 t, 2 dir), 512 threads = 16 waves. Wave: mt = w&3 (16 batch rows),
// ng = w>>2 -> 16 N-tiles, processed in pairs (two independent WMMA chains
// so one chain's pack/store epilogue hides the other's D->VALU hazard).
// =====================================================================
__global__ __launch_bounds__(512) void gates_kernel(const float* __restrict__ x,
                                                    const int* __restrict__ length,
                                                    const unsigned int* __restrict__ wxf,
                                                    const float* __restrict__ biasp,
                                                    unsigned int* __restrict__ gates) {
  int t = blockIdx.x, dir = blockIdx.y;
  int tid = threadIdx.x, wave = tid >> 5, lane = tid & 31;
  int mt = wave & 3, ng = wave >> 2;
  int rowA = mt * 16 + (lane & 15);
  int src = t;
  if (dir) { int L = length[rowA]; src = (t < L) ? (L - 1 - t) : t; }
  const float* xrow = x + ((size_t)rowA * 512 + src) * 256;

  // A fragments: 16x32 bf16 per K-tile, converted from fp32 on the fly
  Frag afr[8];
#pragma unroll
  for (int kt = 0; kt < 8; ++kt) {
    int k0 = kt * 32 + ((lane & 16) >> 1);     // lanes>=16: +8
    f32x4 f0 = *(const f32x4*)(xrow + k0);
    f32x4 f1 = *(const f32x4*)(xrow + k0 + 4);
    f32x4 f2 = *(const f32x4*)(xrow + k0 + 16);
    f32x4 f3 = *(const f32x4*)(xrow + k0 + 20);
#pragma unroll
    for (int i = 0; i < 4; ++i) {
      afr[kt].s[i]      = f2bf(f0[i]);
      afr[kt].s[4 + i]  = f2bf(f1[i]);
      afr[kt].s[8 + i]  = f2bf(f2[i]);
      afr[kt].s[12 + i] = f2bf(f3[i]);
    }
  }

  const unsigned int* wxb = wxf + (size_t)dir * WXDW;
  for (int j = 0; j < 16; j += 2) {
    int nt0 = ng * 16 + j, nt1 = nt0 + 1;
    v8f acc0 = vzero8(), acc1 = vzero8();
#pragma unroll
    for (int kt = 0; kt < 8; ++kt) {
      Frag b0, b1;
      const u32x4* bp0 = (const u32x4*)(wxb + ((size_t)(nt0 * 8 + kt) * 256 + lane * 8));
      const u32x4* bp1 = (const u32x4*)(wxb + ((size_t)(nt1 * 8 + kt) * 256 + lane * 8));
      b0.u4[0] = bp0[0]; b0.u4[1] = bp0[1];
      b1.u4[0] = bp1[0]; b1.u4[1] = bp1[1];
      acc0 = wmma_bf16(afr[kt].bf, b0.bf, acc0);
      acc1 = wmma_bf16(afr[kt].bf, b1.bf, acc1);
    }
    float bias0 = biasp[dir * 1024 + nt0 * 16 + (lane & 15)];
    float bias1 = biasp[dir * 1024 + nt1 * 16 + (lane & 15)];
    u32x4 gs0, gs1;
#pragma unroll
    for (int d = 0; d < 4; ++d) {
      gs0[d] = pack2bf(acc0[2 * d] + bias0, acc0[2 * d + 1] + bias0);
      gs1[d] = pack2bf(acc1[2 * d] + bias1, acc1[2 * d + 1] + bias1);
    }
    size_t base = (((size_t)(dir * 512 + t) * 4 + mt) * 64);
    *(u32x4*)(gates + ((base + nt0) * 128 + lane * 4)) = gs0;
    *(u32x4*)(gates + ((base + nt1) * 128 + lane * 4)) = gs1;
  }
}

// =====================================================================
// K2: persistent recurrent kernel. 2 blocks (dir) x 1024 threads (32 waves).
// W_h bf16 fragments resident in VGPRs (16 frags/wave = 128 VGPRs).
// h DOUBLE-BUFFERED in LDS (2 x 64x256 bf16 = 64 KB): step t reads buf[t&1]
// and writes buf[(t+1)&1], so only ONE barrier per step is needed.
// Each lane keeps its stored h value (hreg) + cell state (c) in registers;
// stores are unconditional so frozen rows (t >= length) persist correctly.
// Per step: issue 8 gates b128 loads + L2 prefetch for t+2; 64 WMMA from
// LDS-A x register-B into zeroed acc; add gates post-GEMM (latency hidden);
// paired swizzle-combine i/g/f/o (one activation chain per lane); store h.
// Wave w owns permuted columns [32w,32w+32) = units [8w,8w+8) x {i,g,f,o}.
// =====================================================================
__global__ __launch_bounds__(1024) void lstm_kernel(const unsigned int* __restrict__ gates,
                                                    const unsigned int* __restrict__ whf,
                                                    const float* __restrict__ cif,
                                                    const float* __restrict__ hif,
                                                    const float* __restrict__ cib,
                                                    const float* __restrict__ hib,
                                                    const int* __restrict__ length,
                                                    unsigned short* __restrict__ hfin) {
  int dir = blockIdx.x;
  int tid = threadIdx.x, wave = tid >> 5, lane = tid & 31;
  __shared__ alignas(16) unsigned short hbuf[2 * 64 * 256];   // 64 KB, double-buffered

  const float* hinit = dir ? hib : hif;
  const float* cinit = dir ? cib : cif;
  for (int idx = tid; idx < 64 * 256; idx += 1024)
    hbuf[idx] = f2bf(hinit[idx & 255]);                        // fill buffer 0

  // resident W_h fragments: N-tiles {2w, 2w+1} x K-tiles 0..7
  Frag bfr[2][8];
#pragma unroll
  for (int n2 = 0; n2 < 2; ++n2)
#pragma unroll
    for (int kt = 0; kt < 8; ++kt) {
      const u32x4* bp = (const u32x4*)(whf + (size_t)dir * WXDW +
                        ((size_t)((2 * wave + n2) * 8 + kt) * 256 + lane * 8));
      bfr[n2][kt].u4[0] = bp[0];
      bfr[n2][kt].u4[1] = bp[1];
    }

  const bool hi8 = (lane & 8) != 0;
  const int rowoff = (lane & 16) >> 1;       // +8 for lanes 16..31 (C layout)
  const int u = wave * 8 + (lane & 7);       // hidden unit owned by this lane
  const int vpar = hi8 ? 1 : 0;              // element parity owned by this lane

  // per-lane state for (row = m*16 + 2j + vpar + rowoff, unit u)
  float c[4][4], hreg[4][4];
  int lenr[4][4];
  float c0 = cinit[u], h0 = hinit[u];
#pragma unroll
  for (int m = 0; m < 4; ++m)
#pragma unroll
    for (int j = 0; j < 4; ++j) {
      c[m][j] = c0;
      hreg[m][j] = h0;
      lenr[m][j] = length[m * 16 + 2 * j + vpar + rowoff];
    }

  __syncthreads();

  for (int t = 0; t < 512; ++t) {
    const int sb = (t & 1) << 14;            // source buffer offset (elements)
    const int db = ((t + 1) & 1) << 14;      // destination buffer offset
    // L2 prefetch of gates fragments for step t+2 (1 KB region per m)
    int t2 = (t + 2 < 512) ? (t + 2) : 511;
#pragma unroll
    for (int m = 0; m < 4; ++m) {
      const char* pf = (const char*)gates +
          ((((size_t)(dir * 512 + t2) * 4 + m) * 64 + 2 * wave) * 128) * 4 + lane * 32;
      __builtin_prefetch((const void*)pf, 0, 1);
    }
    // issue this step's gates fragment loads (consumed after the GEMM)
    u32x4 gld[4][2];
#pragma unroll
    for (int m = 0; m < 4; ++m)
#pragma unroll
      for (int n2 = 0; n2 < 2; ++n2) {
        const u32x4* gp = (const u32x4*)(gates +
            ((((size_t)(dir * 512 + t) * 4 + m) * 64 + 2 * wave + n2) * 128 + lane * 4));
        gld[m][n2] = *gp;
      }
    // h @ W_h into zeroed accumulators
    v8f acc[4][2];
#pragma unroll
    for (int m = 0; m < 4; ++m)
#pragma unroll
      for (int n2 = 0; n2 < 2; ++n2) acc[m][n2] = vzero8();
#pragma unroll
    for (int kt = 0; kt < 8; ++kt) {
      int k0 = kt * 32 + rowoff;             // A layout: lanes>=16 start at K+8
      Frag a[4];
#pragma unroll
      for (int m = 0; m < 4; ++m) {
        int arow = m * 16 + (lane & 15);
        a[m].u4[0] = *(const u32x4*)&hbuf[sb + arow * 256 + k0];
        a[m].u4[1] = *(const u32x4*)&hbuf[sb + arow * 256 + k0 + 16];
      }
#pragma unroll
      for (int m = 0; m < 4; ++m)
#pragma unroll
        for (int n2 = 0; n2 < 2; ++n2)
          acc[m][n2] = wmma_bf16(a[m].bf, bfr[n2][kt].bf, acc[m][n2]);
    }
    // add precomputed x-gates (bf16 fragments -> fp32)
#pragma unroll
    for (int m = 0; m < 4; ++m)
#pragma unroll
      for (int n2 = 0; n2 < 2; ++n2)
#pragma unroll
        for (int d = 0; d < 4; ++d) {
          acc[m][n2][2 * d]     += __uint_as_float(gld[m][n2][d] << 16);
          acc[m][n2][2 * d + 1] += __uint_as_float(gld[m][n2][d] & 0xFFFF0000u);
        }
    // paired elementwise: lo8 lanes own element 2j, hi8 lanes own 2j+1;
    // stores go to the OTHER buffer, so no pre-store barrier is needed.
#pragma unroll
    for (int m = 0; m < 4; ++m)
#pragma unroll
      for (int j = 0; j < 4; ++j) {
        float a00 = acc[m][0][2 * j], a01 = acc[m][0][2 * j + 1];
        float a10 = acc[m][1][2 * j], a11 = acc[m][1][2 * j + 1];
        float s00 = swz8(a00), s01 = swz8(a01);
        float s10 = swz8(a10), s11 = swz8(a11);
        float gi = hi8 ? s01 : a00;   // i of owned element
        float gg = hi8 ? a01 : s00;   // g
        float gf = hi8 ? s11 : a10;   // f
        float go = hi8 ? a11 : s10;   // o
        float cn = sigm(gf) * c[m][j] + sigm(gi) * tanh_f(gg);
        float hv = tanh_f(cn) * sigm(go);
        if (t < lenr[m][j]) { c[m][j] = cn; hreg[m][j] = hv; }
        int row = m * 16 + 2 * j + vpar + rowoff;
        hbuf[db + row * 256 + u] = f2bf(hreg[m][j]);
      }
    __syncthreads();   // writes to buf[(t+1)&1] visible before next step reads
  }

  // final state was written to buffer (511+1)&1 == 0
  for (int idx = tid; idx < 64 * 256; idx += 1024)
    hfin[(size_t)dir * 16384 + idx] = hbuf[idx];
}

// =====================================================================
// K3: out = concat(h_f, h_b) @ W_fc  -> fp32 [64,256]
// 1 block x 512 threads (16 waves); wave: mt = w>>2, 4 N-tiles at (w&3)*4.
// =====================================================================
__global__ __launch_bounds__(512) void out_kernel(const unsigned short* __restrict__ hfin,
                                                  const unsigned int* __restrict__ wfcf,
                                                  float* __restrict__ out) {
  int tid = threadIdx.x, wave = tid >> 5, lane = tid & 31;
  int mt = wave >> 2;
  int j0 = (wave & 3) * 4;
  v8f acc[4];
#pragma unroll
  for (int j = 0; j < 4; ++j) acc[j] = vzero8();
  int rowA = mt * 16 + (lane & 15);
#pragma unroll
  for (int kt = 0; kt < 16; ++kt) {
    int dir = kt >> 3;
    int k0 = (kt & 7) * 32 + ((lane & 16) >> 1);
    const unsigned short* hrow = hfin + (size_t)dir * 16384 + (size_t)rowA * 256;
    Frag a;
    a.u4[0] = *(const u32x4*)(hrow + k0);
    a.u4[1] = *(const u32x4*)(hrow + k0 + 16);
#pragma unroll
    for (int j = 0; j < 4; ++j) {
      int nt = j0 + j;
      Frag b;
      const u32x4* bp = (const u32x4*)(wfcf + ((size_t)(nt * 16 + kt) * 256 + lane * 8));
      b.u4[0] = bp[0]; b.u4[1] = bp[1];
      acc[j] = wmma_bf16(a.bf, b.bf, acc[j]);
    }
  }
  int rowoff = (lane & 16) >> 1;
#pragma unroll
  for (int j = 0; j < 4; ++j) {
    int col = (j0 + j) * 16 + (lane & 15);
#pragma unroll
    for (int v = 0; v < 8; ++v) {
      int row = mt * 16 + v + rowoff;
      out[(size_t)row * 256 + col] = acc[j][v];
    }
  }
}

// =====================================================================
extern "C" void kernel_launch(void* const* d_in, const int* in_sizes, int n_in,
                              void* d_out, int out_size, void* d_ws, size_t ws_size,
                              hipStream_t stream) {
  const float* x      = (const float*)d_in[0];
  const int*   length = (const int*)d_in[1];
  const float* Wf     = (const float*)d_in[2];
  const float* bF     = (const float*)d_in[3];
  const float* Wb     = (const float*)d_in[4];
  const float* bB     = (const float*)d_in[5];
  const float* cif    = (const float*)d_in[6];
  const float* hif    = (const float*)d_in[7];
  const float* cib    = (const float*)d_in[8];
  const float* hib    = (const float*)d_in[9];
  const float* Wfc    = (const float*)d_in[10];
  float* out = (float*)d_out;

  char* ws = (char*)d_ws;   // requires ~130.3 MB of workspace
  unsigned int*   wxf   = (unsigned int*)(ws + OFF_WXF);
  unsigned int*   whf   = (unsigned int*)(ws + OFF_WHF);
  unsigned int*   wfcf  = (unsigned int*)(ws + OFF_WFC);
  float*          biasp = (float*)(ws + OFF_BIAS);
  unsigned short* hfin  = (unsigned short*)(ws + OFF_HFIN);
  unsigned int*   gates = (unsigned int*)(ws + OFF_GATE);

  // K0: weight conversion into WMMA fragment buffers
  prep_kernel<<<2312, 256, 0, stream>>>(Wf, bF, Wb, bB, Wfc, wxf, whf, wfcf, biasp);
  // K1: parallel precompute of x-dependent gate pre-activations (both directions)
  gates_kernel<<<dim3(512, 2), 512, 0, stream>>>(x, length, wxf, biasp, gates);
  // K2: persistent recurrent phase (one workgroup per direction)
  lstm_kernel<<<2, 1024, 0, stream>>>(gates, whf, cif, hif, cib, hib, length, hfin);
  // K3: final projection
  out_kernel<<<1, 512, 0, stream>>>(hfin, wfcf, out);
}
